// LSTM_CNN_CRF_42288247996738
// MI455X (gfx1250) — matrix operand order repaired and checked by reading, not compiled, not run on hardware
//
#include <hip/hip_runtime.h>
#include <hip/hip_bf16.h>

// ---- problem constants (match reference) ----
#define BB   64
#define TT   512
#define LC   16
#define DWD  300
#define CDD  50
#define NFF  100
#define KSS  3
#define HH   256
#define KT   9
#define XD   400     // DW + NF
#define G4   1024    // 4*H
#define MROWS (BB*TT)

typedef __attribute__((ext_vector_type(16))) __bf16          v16bf;
typedef __attribute__((ext_vector_type(8)))  float           v8f;
typedef __attribute__((ext_vector_type(8)))  unsigned short  us8;
typedef int a4i __attribute__((vector_size(4 * sizeof(int))));

union ABFrag { v16bf bf; us8 u[2]; };

#if defined(__AMDGCN__) && __has_builtin(__builtin_amdgcn_global_load_async_to_lds_b128) && __has_builtin(__builtin_amdgcn_s_wait_asynccnt)
#define USE_ASYNC_LDS 1
#else
#define USE_ASYNC_LDS 0
#endif

__device__ __forceinline__ unsigned short f2bf(float f) {
  unsigned u = __builtin_bit_cast(unsigned, f);
  unsigned r = u + 0x7FFFu + ((u >> 16) & 1u);   // round-to-nearest-even
  return (unsigned short)(r >> 16);
}
__device__ __forceinline__ float sigmoidf(float x) {
  return 1.0f / (1.0f + __expf(-x));
}

// ============================================================
// Generic tiled bf16 WMMA GEMM:  Out = A @ Bw^T (+ bias)
//   A:   M x K row-major (f32 or bf16), optional per-length row reversal
//   Bw:  N x K row-major f32 weight
//   Out: row-major (f32 or bf16); optional t-major remap (row = t*B + b)
// Block: 256 thr (8 waves). Block tile 64(M) x 128(N), K step 32.
// Wave w: m-tile = w&3, owns 4 n-tiles at (w>>2)*4 .. +3.
// Staging uses wide b128 paths; only the final partial K segment
// (K=300 case) takes the scalar guarded path.
// ============================================================
__global__ __launch_bounds__(256)
void gemm_bf16_wmma(const void* __restrict__ Aptr, int a_bf16, int lda,
                    const float* __restrict__ Bw, int ldb,
                    const float* __restrict__ bias,
                    void* __restrict__ Out, int out_bf16, int ldo,
                    int M, int N, int Kd,
                    int rev_mode, const int* __restrict__ lengths,
                    int tmajor)
{
  __shared__ unsigned short As[64 * 32];
  __shared__ unsigned short Bs[128 * 32];

  const int tid  = threadIdx.x;
  const int lane = tid & 31;
  const int wv   = tid >> 5;
  const int l16  = lane & 15;
  const int half = lane >> 4;

  const int rowBase = blockIdx.y * 64;
  const int colBase = blockIdx.x * 128;

  const int mt = wv & 3;
  const int ng = wv >> 2;

  v8f acc[4];
  #pragma unroll
  for (int i = 0; i < 4; ++i)
    #pragma unroll
    for (int r = 0; r < 8; ++r) acc[i][r] = 0.0f;

  // --- staging assignments ---
  const int am  = tid >> 2;          // 0..63  A row in tile
  const int aks = (tid & 3) * 8;     // 0,8,16,24
  const int bn  = tid >> 1;          // 0..127 B row (= out col) in tile
  const int bks = (tid & 1) * 16;    // 0,16

  // A source row (with optional length-reversal remap)
  int gr = rowBase + am;
  int b  = gr / TT, t = gr % TT;
  if (rev_mode) { int L = lengths[b]; t = (t < L) ? (L - 1 - t) : t; }
  const long arow = (long)(b * TT + t) * lda;

  const int bgn = colBase + bn;
  const float* __restrict__ bsrc = Bw + (long)bgn * ldb;

  const us8 z8 = {0, 0, 0, 0, 0, 0, 0, 0};

  for (int k0 = 0; k0 < Kd; k0 += 32) {
    __syncthreads();
    // ---- stage A (16B per thread) ----
    {
      unsigned short* dst = &As[am * 32 + aks];
      const int ka = k0 + aks;
      if (ka + 8 <= Kd) {
        if (a_bf16) {
          const unsigned short* src = (const unsigned short*)Aptr + arow + ka;
#if USE_ASYNC_LDS
          __builtin_amdgcn_global_load_async_to_lds_b128(
              (a4i*)src,
              (__attribute__((address_space(3))) a4i*)dst, 0, 0);
#else
          *(us8*)dst = *(const us8*)src;
#endif
        } else {
          const float4 v0 = *(const float4*)((const float*)Aptr + arow + ka);
          const float4 v1 = *(const float4*)((const float*)Aptr + arow + ka + 4);
          us8 o;
          o[0] = f2bf(v0.x); o[1] = f2bf(v0.y); o[2] = f2bf(v0.z); o[3] = f2bf(v0.w);
          o[4] = f2bf(v1.x); o[5] = f2bf(v1.y); o[6] = f2bf(v1.z); o[7] = f2bf(v1.w);
          *(us8*)dst = o;
        }
      } else if (ka >= Kd) {
        *(us8*)dst = z8;
      } else {                 // partial segment (only last chunk, K=300)
        #pragma unroll
        for (int j = 0; j < 8; ++j) {
          const int k = ka + j;
          unsigned short v = 0;
          if (k < Kd) {
            v = a_bf16 ? ((const unsigned short*)Aptr)[arow + k]
                       : f2bf(((const float*)Aptr)[arow + k]);
          }
          dst[j] = v;
        }
      }
    }
    // ---- stage B (16 f32 -> 16 bf16 per thread) ----
    {
      unsigned short* dst = &Bs[bn * 32 + bks];
      const int kb = k0 + bks;
      if (bgn < N && kb + 16 <= Kd) {
        const float4 w0 = *(const float4*)(bsrc + kb);
        const float4 w1 = *(const float4*)(bsrc + kb + 4);
        const float4 w2 = *(const float4*)(bsrc + kb + 8);
        const float4 w3 = *(const float4*)(bsrc + kb + 12);
        us8 o0, o1;
        o0[0] = f2bf(w0.x); o0[1] = f2bf(w0.y); o0[2] = f2bf(w0.z); o0[3] = f2bf(w0.w);
        o0[4] = f2bf(w1.x); o0[5] = f2bf(w1.y); o0[6] = f2bf(w1.z); o0[7] = f2bf(w1.w);
        o1[0] = f2bf(w2.x); o1[1] = f2bf(w2.y); o1[2] = f2bf(w2.z); o1[3] = f2bf(w2.w);
        o1[4] = f2bf(w3.x); o1[5] = f2bf(w3.y); o1[6] = f2bf(w3.z); o1[7] = f2bf(w3.w);
        *(us8*)dst       = o0;
        *(us8*)(dst + 8) = o1;
      } else if (bgn >= N || kb >= Kd) {
        *(us8*)dst       = z8;
        *(us8*)(dst + 8) = z8;
      } else {                 // partial segment (only last chunk, K=300)
        #pragma unroll
        for (int j = 0; j < 16; ++j) {
          const int k = kb + j;
          dst[j] = (k < Kd) ? f2bf(bsrc[k]) : (unsigned short)0;
        }
      }
    }
    // prefetch next B chunk into caches (global_prefetch_b8)
    if (k0 + 32 < Kd && bgn < N)
      __builtin_prefetch((const char*)(bsrc + k0 + 32), 0, 3);
#if USE_ASYNC_LDS
    __builtin_amdgcn_s_wait_asynccnt(0);
#endif
    __syncthreads();

    // ---- fragments + WMMA ----
    ABFrag a;
    {
      const int ar = (mt * 16 + l16) * 32;
      const int c0 = half ? 8 : 0;
      a.u[0] = *(const us8*)&As[ar + c0];
      a.u[1] = *(const us8*)&As[ar + c0 + 16];
    }
    #pragma unroll
    for (int nt = 0; nt < 4; ++nt) {
      ABFrag bf;
      const int br = ((ng * 4 + nt) * 16 + l16) * 32 + (half ? 8 : 0);
      bf.u[0] = *(const us8*)&Bs[br];
      bf.u[1] = *(const us8*)&Bs[br + 16];
      acc[nt] = __builtin_amdgcn_wmma_f32_16x16x32_bf16(
          false, a.bf, false, bf.bf, (short)0, acc[nt], false, false);
    }
  }

  // ---- store (C layout: lane<16 -> M=r, lane>=16 -> M=r+8; N = lane&15) ----
  #pragma unroll
  for (int nt = 0; nt < 4; ++nt) {
    const int col = colBase + (ng * 4 + nt) * 16 + l16;
    if (col >= N) continue;
    const float bv = bias ? bias[col] : 0.0f;
    #pragma unroll
    for (int r = 0; r < 8; ++r) {
      const int gr2 = rowBase + mt * 16 + half * 8 + r;
      long orow;
      if (tmajor) { int b2 = gr2 / TT, t2 = gr2 % TT; orow = (long)(t2 * BB + b2) * ldo; }
      else        { orow = (long)gr2 * ldo; }
      const float v = acc[nt][r] + bv;
      if (out_bf16) ((unsigned short*)Out)[orow + col] = f2bf(v);
      else          ((float*)Out)[orow + col] = v;
    }
  }
}

// ============================================================
// f32 -> bf16 elementwise convert
// ============================================================
__global__ void cvt_bf16(const float* __restrict__ src,
                         unsigned short* __restrict__ dst, int n)
{
  int i = blockIdx.x * blockDim.x + threadIdx.x;
  if (i < n) dst[i] = f2bf(src[i]);
}

// ============================================================
// Char CNN: per (b,t): gather 16 char embeddings, conv(K=3, SAME),
// relu, max over positions -> x[:, 300:400] (bf16)
// Block 256 thr = 8 waves, one wave per (b,t) position.
// ============================================================
__global__ __launch_bounds__(256)
void char_cnn(const int* __restrict__ chars, const float* __restrict__ cemb,
              const float* __restrict__ convW, const float* __restrict__ convb,
              unsigned short* __restrict__ x)
{
  __shared__ float ce[8][LC * CDD];
  const int tid = threadIdx.x, lane = tid & 31, wv = tid >> 5;
  const long pos = (long)blockIdx.x * 8 + wv;

  const int* cp = chars + pos * LC;
  for (int j = lane; j < LC * CDD; j += 32) {
    int p = j / CDD, c = j % CDD;
    ce[wv][j] = cemb[cp[p] * CDD + c];
  }
  __syncthreads();

  int   nq[4];
  float bq[4], mx[4];
  #pragma unroll
  for (int q = 0; q < 4; ++q) {
    nq[q] = lane + q * 32;
    bq[q] = (nq[q] < NFF) ? convb[nq[q]] : 0.0f;
    mx[q] = 0.0f;                      // relu >= 0
  }

  for (int p = 0; p < LC; ++p) {
    float aq[4];
    #pragma unroll
    for (int q = 0; q < 4; ++q) aq[q] = bq[q];
    for (int dk = 0; dk < KSS; ++dk) {
      const int pp = p + dk - 1;
      if (pp < 0 || pp >= LC) continue;
      const float* cerow = &ce[wv][pp * CDD];
      #pragma unroll 1
      for (int c = 0; c < CDD; ++c) {
        const float cv = cerow[c];
        const float* wrow = convW + (dk * CDD + c) * NFF;
        #pragma unroll
        for (int q = 0; q < 4; ++q)
          if (nq[q] < NFF) aq[q] += cv * wrow[nq[q]];
      }
    }
    #pragma unroll
    for (int q = 0; q < 4; ++q) {
      float r = aq[q] > 0.0f ? aq[q] : 0.0f;
      mx[q] = fmaxf(mx[q], r);
    }
  }
  #pragma unroll
  for (int q = 0; q < 4; ++q)
    if (nq[q] < NFF) x[pos * XD + DWD + nq[q]] = f2bf(mx[q]);
}

// ============================================================
// LSTM recurrence, one direction. Single workgroup, 1024 thr = 32 waves.
// h state (64x256 bf16) lives in LDS; c/h fragments live in registers.
// Wave w: m-tile = w&3 (16 batch rows), hidden tiles 2*(w>>2), +1.
// Per hidden tile it owns all 4 gate output tiles -> cell in registers.
// ============================================================
__global__ __launch_bounds__(1024)
void lstm_rec(const float* __restrict__ pre,          // (T, B, 1024)
              const unsigned short* __restrict__ Whh, // (1024, 256) bf16
              const int* __restrict__ lengths,
              float* __restrict__ out,                // (B, T, 512)
              int coloff, int rev)
{
  __shared__ unsigned short hbf[BB * HH];   // 32 KB

  const int tid  = threadIdx.x;
  const int lane = tid & 31;
  const int wv   = tid >> 5;
  const int l16  = lane & 15;
  const int half = lane >> 4;
  const int mt   = wv & 3;
  const int hg   = wv >> 2;                 // 0..7

  for (int i = tid; i < BB * HH; i += 1024) hbf[i] = 0;

  v8f cst[2], hst[2];
  #pragma unroll
  for (int u = 0; u < 2; ++u)
    #pragma unroll
    for (int r = 0; r < 8; ++r) { cst[u][r] = 0.0f; hst[u][r] = 0.0f; }

  int brow[8], lenb[8];
  #pragma unroll
  for (int r = 0; r < 8; ++r) {
    brow[r] = mt * 16 + half * 8 + r;       // batch index (C-layout row)
    lenb[r] = lengths[brow[r]];
  }
  __syncthreads();

  for (int s = 0; s < TT; ++s) {
    #pragma unroll
    for (int u = 0; u < 2; ++u) {
      const int ht = hg * 2 + u;
      v8f g[4];
      #pragma unroll
      for (int gi = 0; gi < 4; ++gi) {
        const int ncol = gi * HH + ht * 16 + l16;
        v8f a;
        #pragma unroll
        for (int r = 0; r < 8; ++r)
          a[r] = pre[((long)s * BB + brow[r]) * G4 + ncol];
        const unsigned short* wrow = Whh + (long)ncol * HH;
        #pragma unroll
        for (int kc = 0; kc < 8; ++kc) {
          ABFrag af;  // A = h (M=batch, K=hidden), from LDS
          {
            const int ar = (mt * 16 + l16) * HH + kc * 32 + (half ? 8 : 0);
            af.u[0] = *(const us8*)&hbf[ar];
            af.u[1] = *(const us8*)&hbf[ar + 16];
          }
          ABFrag bf;  // B = Whh^T column = Whh row ncol
          {
            const int off = kc * 32 + (half ? 8 : 0);
            bf.u[0] = *(const us8*)&wrow[off];
            bf.u[1] = *(const us8*)&wrow[off + 16];
          }
          a = __builtin_amdgcn_wmma_f32_16x16x32_bf16(
              false, af.bf, false, bf.bf, (short)0, a, false, false);
        }
        g[gi] = a;
      }
      // LSTM cell (registers only; LDS writes deferred past barrier)
      #pragma unroll
      for (int r = 0; r < 8; ++r) {
        const float iv = sigmoidf(g[0][r]);
        const float fv = sigmoidf(g[1][r]);
        const float gv = tanhf(g[2][r]);
        const float ov = sigmoidf(g[3][r]);
        const float cn = fv * cst[u][r] + iv * gv;
        const float hn = ov * tanhf(cn);
        const bool  mk = (s < lenb[r]);
        hst[u][r] = mk ? hn : hst[u][r];
        cst[u][r] = mk ? cn : cst[u][r];
      }
    }
    __syncthreads();   // all LDS h reads of this step are done
    #pragma unroll
    for (int u = 0; u < 2; ++u) {
      const int ht = hg * 2 + u;
      const int nidx = ht * 16 + l16;
      #pragma unroll
      for (int r = 0; r < 8; ++r) {
        hbf[brow[r] * HH + nidx] = f2bf(hst[u][r]);
        const bool mk = (s < lenb[r]);
        const int tout = rev ? (mk ? (lenb[r] - 1 - s) : s) : s;
        out[((long)brow[r] * TT + tout) * (2 * HH) + coloff + nidx] =
            mk ? hst[u][r] : 0.0f;
      }
    }
    __syncthreads();   // new h visible for next step
  }
}

// ============================================================
// logits = lstm_out @ out_W + out_b   (N=9, plain VALU)
// ============================================================
__global__ void logits_k(const float* __restrict__ lo,
                         const float* __restrict__ outW,
                         const float* __restrict__ outb,
                         float* __restrict__ lg)
{
  long i = (long)blockIdx.x * blockDim.x + threadIdx.x;
  if (i >= (long)MROWS * KT) return;
  const int  k   = (int)(i % KT);
  const long row = i / KT;
  float s = outb[k];
  const float* r = lo + row * (2 * HH);
  for (int c = 0; c < 2 * HH; ++c) s += r[c] * outW[c * KT + k];
  lg[i] = s;
}

// ============================================================
// CRF negative log-likelihood (single block, 64*9 = 576 threads)
// ============================================================
__global__ __launch_bounds__(576)
void crf_loss(const float* __restrict__ lg, const int* __restrict__ tags,
              const int* __restrict__ lengths, const float* __restrict__ trans,
              const float* __restrict__ stt, const float* __restrict__ ent,
              float* __restrict__ loss_out)
{
  __shared__ float alf[BB][KT];
  __shared__ float trs[KT * KT], st[KT], en[KT];
  __shared__ float zb[BB], scb[BB];
  const int tid = threadIdx.x;
  if (tid < KT * KT) trs[tid] = trans[tid];
  if (tid < KT) { st[tid] = stt[tid]; en[tid] = ent[tid]; }
  __syncthreads();

  const int b = tid / KT, k = tid % KT;
  float a = st[k] + lg[((long)b * TT) * KT + k];
  alf[b][k] = a;
  for (int t = 1; t < TT; ++t) {
    __syncthreads();
    float arow[KT];
    #pragma unroll
    for (int j = 0; j < KT; ++j) arow[j] = alf[b][j];
    float mv = -1e30f, v[KT];
    #pragma unroll
    for (int j = 0; j < KT; ++j) {
      v[j] = arow[j] + trs[j * KT + k];
      mv = fmaxf(mv, v[j]);
    }
    float ss = 0.0f;
    #pragma unroll
    for (int j = 0; j < KT; ++j) ss += __expf(v[j] - mv);
    const float nxt = mv + __logf(ss) + lg[((long)b * TT + t) * KT + k];
    a = (t < lengths[b]) ? nxt : a;
    __syncthreads();
    alf[b][k] = a;
  }
  __syncthreads();
  if (k == 0) {
    float mv = -1e30f;
    for (int j = 0; j < KT; ++j) mv = fmaxf(mv, alf[b][j] + en[j]);
    float ss = 0.0f;
    for (int j = 0; j < KT; ++j) ss += __expf(alf[b][j] + en[j] - mv);
    zb[b] = mv + __logf(ss);
    const int len = lengths[b];
    int prev = tags[b * TT];
    float sc = st[prev] + lg[((long)b * TT) * KT + prev];
    for (int t = 1; t < TT; ++t) {
      const int tg = tags[b * TT + t];
      if (t < len) {
        sc += trs[prev * KT + tg] + lg[((long)b * TT + t) * KT + tg];
        prev = tg;
      }
    }
    scb[b] = sc + en[prev];
  }
  __syncthreads();
  if (tid == 0) {
    float L = 0.0f;
    for (int bb = 0; bb < BB; ++bb) L += zb[bb] - scb[bb];
    loss_out[0] = L;    // loss = -(score - Z) summed
  }
}

// ============================================================
// Viterbi decode (single block, 576 threads) -> preds as floats
// ============================================================
__global__ __launch_bounds__(576)
void viterbi_k(const float* __restrict__ lg, const int* __restrict__ lengths,
               const float* __restrict__ trans, const float* __restrict__ stt,
               const float* __restrict__ ent, int* __restrict__ bp,
               float* __restrict__ preds)
{
  __shared__ float alf[BB][KT];
  __shared__ float trs[KT * KT];
  const int tid = threadIdx.x;
  if (tid < KT * KT) trs[tid] = trans[tid];
  __syncthreads();

  const int b = tid / KT, k = tid % KT;
  float a = stt[k] + lg[((long)b * TT) * KT + k];
  alf[b][k] = a;
  for (int t = 1; t < TT; ++t) {
    __syncthreads();
    float arow[KT];
    #pragma unroll
    for (int j = 0; j < KT; ++j) arow[j] = alf[b][j];
    float bv = -1e30f; int bj = 0;
    #pragma unroll
    for (int j = 0; j < KT; ++j) {
      const float c = arow[j] + trs[j * KT + k];
      if (c > bv) { bv = c; bj = j; }
    }
    const float nxt = bv + lg[((long)b * TT + t) * KT + k];
    const bool mk = (t < lengths[b]);
    bp[((long)t * BB + b) * KT + k] = mk ? bj : k;
    a = mk ? nxt : a;
    __syncthreads();
    alf[b][k] = a;
  }
  __syncthreads();
  if (k == 0) {
    int tag = 0; float bv = -1e30f;
    for (int j = 0; j < KT; ++j) {
      const float v = alf[b][j] + ent[j];
      if (v > bv) { bv = v; tag = j; }
    }
    for (int t = TT - 1; t >= 1; --t) {
      preds[b * TT + t] = (t < lengths[b]) ? (float)tag : 0.0f;
      tag = bp[((long)t * BB + b) * KT + tag];
    }
    preds[b * TT + 0] = (float)tag;   // lengths >= T/2 > 0
  }
}

// ============================================================
extern "C" void kernel_launch(void* const* d_in, const int* in_sizes, int n_in,
                              void* d_out, int out_size, void* d_ws, size_t ws_size,
                              hipStream_t stream) {
  const float* word_embs  = (const float*)d_in[0];
  const int*   chars      = (const int*)  d_in[1];
  const int*   lengths    = (const int*)  d_in[2];
  const int*   tags       = (const int*)  d_in[3];
  const float* fin_W      = (const float*)d_in[4];
  const float* fin_b      = (const float*)d_in[5];
  const float* char_emb   = (const float*)d_in[6];
  const float* conv_W     = (const float*)d_in[7];
  const float* conv_b     = (const float*)d_in[8];
  const float* Wih_f      = (const float*)d_in[9];
  const float* Whh_f      = (const float*)d_in[10];
  const float* b_f        = (const float*)d_in[11];
  const float* Wih_b      = (const float*)d_in[12];
  const float* Whh_b      = (const float*)d_in[13];
  const float* b_b        = (const float*)d_in[14];
  const float* out_W      = (const float*)d_in[15];
  const float* out_b      = (const float*)d_in[16];
  const float* trans      = (const float*)d_in[17];
  const float* start_tr   = (const float*)d_in[18];
  const float* end_tr     = (const float*)d_in[19];

  // ---- workspace carving ----
  size_t off = 0;
  char* ws = (char*)d_ws;
  auto carve = [&](size_t bytes) -> void* {
    void* p = ws + off;
    off += (bytes + 255) & ~(size_t)255;
    return p;
  };
  unsigned short* x_bf  = (unsigned short*)carve((size_t)MROWS * XD * 2);
  unsigned short* whhf  = (unsigned short*)carve((size_t)G4 * HH * 2);
  unsigned short* whhb  = (unsigned short*)carve((size_t)G4 * HH * 2);
  float* pre_f  = (float*)carve((size_t)TT * BB * G4 * 4);
  float* pre_b  = (float*)carve((size_t)TT * BB * G4 * 4);
  float* lstm_o = (float*)carve((size_t)MROWS * 2 * HH * 4);
  float* lgts   = (float*)carve((size_t)MROWS * KT * 4);
  int*   bp     = (int*)  carve((size_t)TT * BB * KT * 4);

  // 1) Whh f32 -> bf16
  cvt_bf16<<<(G4 * HH + 255) / 256, 256, 0, stream>>>(Whh_f, whhf, G4 * HH);
  cvt_bf16<<<(G4 * HH + 255) / 256, 256, 0, stream>>>(Whh_b, whhb, G4 * HH);

  // 2) word projection -> x[:, 0:300] (bf16)
  {
    dim3 g((DWD + 127) / 128, MROWS / 64);
    gemm_bf16_wmma<<<g, 256, 0, stream>>>(word_embs, 0, DWD, fin_W, DWD, fin_b,
                                          x_bf, 1, XD, MROWS, DWD, DWD,
                                          0, nullptr, 0);
  }
  // 3) char CNN -> x[:, 300:400] (bf16)
  char_cnn<<<MROWS / 8, 256, 0, stream>>>(chars, char_emb, conv_W, conv_b, x_bf);

  // 4) input projections (t-major pre), backward reads length-reversed rows
  {
    dim3 g(G4 / 128, MROWS / 64);
    gemm_bf16_wmma<<<g, 256, 0, stream>>>(x_bf, 1, XD, Wih_f, XD, b_f,
                                          pre_f, 0, G4, MROWS, G4, XD,
                                          0, lengths, 1);
    gemm_bf16_wmma<<<g, 256, 0, stream>>>(x_bf, 1, XD, Wih_b, XD, b_b,
                                          pre_b, 0, G4, MROWS, G4, XD,
                                          1, lengths, 1);
  }

  // 5) recurrences (fwd writes cols 0:256, bwd writes cols 256:512 un-reversed)
  lstm_rec<<<1, 1024, 0, stream>>>(pre_f, whhf, lengths, lstm_o, 0, 0);
  lstm_rec<<<1, 1024, 0, stream>>>(pre_b, whhb, lengths, lstm_o, HH, 1);

  // 6) logits
  {
    const long n = (long)MROWS * KT;
    logits_k<<<(unsigned)((n + 255) / 256), 256, 0, stream>>>(lstm_o, out_W,
                                                              out_b, lgts);
  }

  // 7) CRF loss -> d_out[0];  8) Viterbi -> d_out[1..]
  crf_loss<<<1, 576, 0, stream>>>(lgts, tags, lengths, trans, start_tr, end_tr,
                                  (float*)d_out);
  viterbi_k<<<1, 576, 0, stream>>>(lgts, lengths, trans, start_tr, end_tr, bp,
                                   (float*)d_out + 1);
}
